// SpatiallyAdaptiveConv_36636071035630
// MI455X (gfx1250) — matrix operand order repaired and checked by reading, not compiled, not run on hardware
//
#include <hip/hip_runtime.h>
#include <hip/hip_bf16.h>
#include <cstdint>

// ---------------------------------------------------------------------------
// Types for WMMA / TDM builtins
// ---------------------------------------------------------------------------
typedef __attribute__((ext_vector_type(16))) __bf16        v16bf;
typedef __attribute__((ext_vector_type(8)))  float         v8f;
typedef __attribute__((ext_vector_type(4)))  unsigned int  u32x4;
typedef __attribute__((ext_vector_type(8)))  int           i32x8;
typedef __attribute__((ext_vector_type(4)))  int           i32x4;

// round-to-nearest-even f32 -> bf16 (bit pattern as ushort)
static __device__ inline unsigned short f2bf(float f) {
    unsigned u = __builtin_bit_cast(unsigned, f);
    unsigned r = 0x7fffu + ((u >> 16) & 1u);
    return (unsigned short)((u + r) >> 16);
}

// 32-byte LDS fragment assembly (two ds_load_b128)
struct Q32 { uint4 a; uint4 b; };
static __device__ inline v16bf load_frag(const unsigned short* p0,
                                         const unsigned short* p1) {
    Q32 q;
    q.a = *reinterpret_cast<const uint4*>(p0);
    q.b = *reinterpret_cast<const uint4*>(p1);
    return __builtin_bit_cast(v16bf, q);
}

// ---------------------------------------------------------------------------
// Tensor Data Mover: 2D tile (bf16 elements) global -> LDS.
// D# layout per CDNA5 ISA ch.8 (group0 128b, group1 256b; groups 2/3 unused).
// ---------------------------------------------------------------------------
static __device__ inline void tdm_load_2d_bf16(unsigned lds_off, const void* gptr,
                                               unsigned tensor_d0, unsigned tensor_d1,
                                               unsigned tile_d0, unsigned tile_d1,
                                               unsigned long long stride0_elems) {
    unsigned long long ga = (unsigned long long)(uintptr_t)gptr;
    u32x4 g0;
    g0[0] = 1u;                                   // count=1, user descriptor
    g0[1] = lds_off;                              // LDS byte address
    g0[2] = (unsigned)ga;                         // global_addr[31:0]  (D# bits 95:64)
    g0[3] = (unsigned)((ga >> 32) & 0x1ffffffull) // global_addr[56:32]
          | (2u << 30);                           // type = 2 ("image")

    i32x8 g1;
    g1[0] = (int)(1u << 16);                                            // data_size=1 (2B); wg_mask=0
    g1[1] = (int)((tensor_d0 & 0xffffu) << 16);                         // tensor_dim0[15:0]
    g1[2] = (int)((tensor_d0 >> 16) | ((tensor_d1 & 0xffffu) << 16));   // dim0[31:16] | dim1[15:0]
    g1[3] = (int)((tensor_d1 >> 16) | (tile_d0 << 16));                 // dim1[31:16] | tile_dim0
    g1[4] = (int)(tile_d1 & 0xffffu);                                   // tile_dim1 (tile_dim2=0)
    g1[5] = (int)(unsigned)(stride0_elems & 0xffffffffull);             // dim0_stride[31:0]
    g1[6] = (int)(unsigned)((stride0_elems >> 32) & 0xffffull);         // dim0_stride[47:32]
    g1[7] = 0;

    i32x4 z4 = {0, 0, 0, 0};
#if __clang_major__ >= 23
    i32x8 z8 = {0, 0, 0, 0, 0, 0, 0, 0};
    __builtin_amdgcn_tensor_load_to_lds(g0, g1, z4, z4, z8, 0);
#else
    __builtin_amdgcn_tensor_load_to_lds(g0, g1, z4, z4, 0);
#endif
}

// ---------------------------------------------------------------------------
// Prep kernels
// ---------------------------------------------------------------------------
// h = relu(W1 @ X + b1); stored transposed (HT[p][o], K-contiguous) for B-side TDM
__global__ __launch_bounds__(256) void prep_h(const float* __restrict__ x,
                                              const float* __restrict__ w1,
                                              const float* __restrict__ b1,
                                              unsigned short* __restrict__ HT) {
    int t = blockIdx.x * 256 + threadIdx.x;      // 65536 = 64 * 1024
    int o = t >> 10, p = t & 1023;
    float acc = b1[o];
    #pragma unroll 8
    for (int c = 0; c < 64; ++c) acc = fmaf(w1[o * 64 + c], x[c * 1024 + p], acc);
    acc = fmaxf(acc, 0.0f);
    HT[p * 64 + o] = f2bf(acc);
}

// im2col(x), transposed: XT[p][r], r = cin*9 + kh*3 + kw, K-contiguous (bf16)
__global__ __launch_bounds__(256) void prep_xcol(const float* __restrict__ x,
                                                 unsigned short* __restrict__ XT) {
    int t = blockIdx.x * 256 + threadIdx.x;      // 589824 = 1024 * 576
    int p = t / 576, r = t - p * 576;
    int cin = r / 9, rr = r - cin * 9;
    int dy = rr / 3 - 1, dx = rr - (rr / 3) * 3 - 1;
    int y = (p >> 5) + dy, xx = (p & 31) + dx;
    float v = (y >= 0 && y < 32 && xx >= 0 && xx < 32) ? x[cin * 1024 + y * 32 + xx] : 0.0f;
    XT[t] = f2bf(v);
}

// w2 f32 -> bf16 (row-major [18432][64])
__global__ __launch_bounds__(256) void prep_w2(const float* __restrict__ w2,
                                               unsigned short* __restrict__ w2b) {
    int t = blockIdx.x * 256 + threadIdx.x;      // 1179648
    w2b[t] = f2bf(w2[t]);
}

// ---------------------------------------------------------------------------
// bf16 WMMA GEMM:  OUT[M x 1024] = A[M x KTOT] * B^T  (B stored [1024][KTOT])
// Block tile 128x64, 8 waves (each 16x64), K-step 32, TDM double buffering.
// ---------------------------------------------------------------------------
template <int KTOT, int MTOT, bool HAS_BIAS, bool OUT_BF16>
__global__ __launch_bounds__(256) void gemm_wmma(const unsigned short* __restrict__ A,
                                                 const unsigned short* __restrict__ B,
                                                 const float* __restrict__ bias,
                                                 void* __restrict__ outp) {
    constexpr int KSTEPS = KTOT / 32;
    __shared__ unsigned short Abuf[2][128 * 32];   // rows m, K-contiguous (64B rows)
    __shared__ unsigned short Bbuf[2][64 * 32];    // cols n, K-contiguous (64B rows)

    const int m0   = blockIdx.x * 128;
    const int n0   = blockIdx.y * 64;
    const int tid  = threadIdx.x;
    const int wave = tid >> 5;
    const int lane = tid & 31;
    const int lo16 = lane & 15;
    const int hi   = lane >> 4;
    const bool wave0 = (wave == 0);

    auto issue = [&](int buf, int k0) {
        tdm_load_2d_bf16((unsigned)(uintptr_t)&Abuf[buf][0],
                         A + (size_t)m0 * KTOT + k0,
                         (unsigned)KTOT, (unsigned)MTOT, 32u, 128u,
                         (unsigned long long)KTOT);
        tdm_load_2d_bf16((unsigned)(uintptr_t)&Bbuf[buf][0],
                         B + (size_t)n0 * KTOT + k0,
                         (unsigned)KTOT, 1024u, 32u, 64u,
                         (unsigned long long)KTOT);
    };

    if (wave0) issue(0, 0);

    v8f c[4] = {};

    for (int s = 0; s < KSTEPS; ++s) {
        const int cur = s & 1;
        if (s + 1 < KSTEPS) {
            if (wave0) issue(cur ^ 1, (s + 1) * 32);
            __builtin_amdgcn_s_wait_tensorcnt(2);   // the two loads for `cur` done
        } else {
            __builtin_amdgcn_s_wait_tensorcnt(0);
        }
        __syncthreads();

        // A fragment 16x32 bf16 per ISA striping:
        // lanes 0-15: K 0-7 then 16-23 ; lanes 16-31: K 8-15 then 24-31
        const unsigned short* abase = &Abuf[cur][(wave * 16 + lo16) * 32];
        const v16bf afrag = load_frag(abase + (hi ? 8 : 0),
                                      abase + 16 + (hi ? 8 : 0));

        // 4 N-subtiles of 16: B fragment 32x16, lane = column, K striped like A
        #pragma unroll
        for (int j = 0; j < 4; ++j) {
            const unsigned short* bbase = &Bbuf[cur][(j * 16 + lo16) * 32 + (hi ? 8 : 0)];
            const v16bf bfrag = load_frag(bbase, bbase + 16);
            c[j] = __builtin_amdgcn_wmma_f32_16x16x32_bf16(
                       false, afrag, false, bfrag, (short)0, c[j], false, false);
        }
        __syncthreads();
    }

    // Epilogue: C/D layout — VGPR r: lanes 0-15 -> M=r, lanes 16-31 -> M=8+r; N=lane&15
    #pragma unroll
    for (int r = 0; r < 8; ++r) {
        const int row = m0 + wave * 16 + hi * 8 + r;
        float bv = 0.0f;
        if constexpr (HAS_BIAS) bv = bias[row];
        #pragma unroll
        for (int j = 0; j < 4; ++j) {
            const int col = n0 + j * 16 + lo16;
            const float v = c[j][r] + bv;
            if constexpr (OUT_BF16)
                ((unsigned short*)outp)[(size_t)row * 1024 + col] = f2bf(v);
            else
                ((float*)outp)[(size_t)row * 1024 + col] = v;
        }
    }
}

// ---------------------------------------------------------------------------
// Host launch
// ---------------------------------------------------------------------------
extern "C" void kernel_launch(void* const* d_in, const int* in_sizes, int n_in,
                              void* d_out, int out_size, void* d_ws, size_t ws_size,
                              hipStream_t stream) {
    const float* x  = (const float*)d_in[0];   // [64,32,32]
    const float* w1 = (const float*)d_in[1];   // [64,64]
    const float* b1 = (const float*)d_in[2];   // [64]
    const float* w2 = (const float*)d_in[3];   // [18432,64]
    const float* b2 = (const float*)d_in[4];   // [18432]
    float* out = (float*)d_out;                // [32768,1024]

    char* ws = (char*)d_ws;                    // needs ~41.4 MB scratch
    unsigned short* HT  = (unsigned short*)(ws + 0);        // [1024][64]   128 KB
    unsigned short* XT  = (unsigned short*)(ws + 131072);   // [1024][576] 1.13 MB
    unsigned short* W2B = (unsigned short*)(ws + 1310720);  // [18432][64] 2.25 MB
    unsigned short* AWB = (unsigned short*)(ws + 3670016);  // [18432][1024] == [32768][576], 36 MB

    prep_h   <<<256,  256, 0, stream>>>(x, w1, b1, HT);
    prep_xcol<<<2304, 256, 0, stream>>>(x, XT);
    prep_w2  <<<4608, 256, 0, stream>>>(w2, W2B);

    // Stage 2: AW = W2 * H + b2  (M=18432, N=1024, K=64), bf16 out
    gemm_wmma<64, 18432, true, true>
        <<<dim3(144, 16), 256, 0, stream>>>(W2B, HT, b2, AWB);

    // Stage 3: OUT = Wconv * im2col(X)  (M=32768, N=1024, K=576), f32 out
    gemm_wmma<576, 32768, false, false>
        <<<dim3(256, 16), 256, 0, stream>>>(AWB, XT, nullptr, out);
}
